// MHAtt_61040075210973
// MI455X (gfx1250) — compile-verified, matrix operand order checked by reading, not run on hardware
//
#include <hip/hip_runtime.h>

#define HSZ   1024
#define NH    16
#define DH    64
#define BATCH 2
#define SEQ   2048
#define MROWS (BATCH*SEQ)          // 4096

typedef __attribute__((ext_vector_type(16))) __bf16        bf16x16;
typedef __attribute__((ext_vector_type(8)))  float         f32x8;
typedef __attribute__((ext_vector_type(4)))  unsigned int  u32x4;

union FragU {
    bf16x16 v;
    u32x4   q[2];
};

__device__ __forceinline__ unsigned short f32_to_bf16(float f) {
    unsigned int u = __float_as_uint(f);
    u += 0x7FFFu + ((u >> 16) & 1u);           // round-to-nearest-even
    return (unsigned short)(u >> 16);
}

// gfx1250 async global->LDS copy, 16 bytes per lane, tracked by ASYNCcnt.
// Low 32 bits of a generic pointer into __shared__ are the LDS byte offset.
__device__ __forceinline__ void async_copy_b128(const void* gsrc, void* ldst) {
    unsigned lds            = (unsigned)(uintptr_t)ldst;
    unsigned long long gadr = (unsigned long long)(uintptr_t)gsrc;
    asm volatile("global_load_async_to_lds_b128 %0, %1, off"
                 :: "v"(lds), "v"(gadr) : "memory");
}

__device__ __forceinline__ void wait_async_zero() {
#if __has_builtin(__builtin_amdgcn_s_wait_asynccnt)
    __builtin_amdgcn_s_wait_asynccnt(0);
#else
    asm volatile("s_wait_asynccnt 0" ::: "memory");
#endif
}

// ---------------------------------------------------------------- convert ---
__global__ void cvt_bf16_kernel(const float* __restrict__ in,
                                unsigned short* __restrict__ out, int n) {
    for (int i = blockIdx.x * blockDim.x + threadIdx.x; i < n;
         i += gridDim.x * blockDim.x)
        out[i] = f32_to_bf16(in[i]);
}

// ------------------------------------------------------------------- GEMM ---
// C[m,n] = sum_k A[m,k] * B[n,k] + bias[n]   (A: [4096,1024] bf16, B: [1024,1024] bf16)
// mode 0: f32 out, row-major [MROWS, HSZ]
// mode 1: bf16 out, head-split [B][NH][SEQ][DH]
// mode 2: bf16 out, transposed [B][NH][DH][SEQ]
__global__ __launch_bounds__(128)
void gemm_bf16_kernel(const unsigned short* __restrict__ A,
                      const unsigned short* __restrict__ Bm,
                      const float* __restrict__ bias,
                      void* __restrict__ out, int mode) {
    __shared__ unsigned short As[64 * 32];
    __shared__ unsigned short Bs[64 * 32];

    const int n0   = blockIdx.x * 64;
    const int m0   = blockIdx.y * 64;
    const int tid  = threadIdx.x;
    const int lane = tid & 31;
    const int wave = tid >> 5;
    const int wm   = (wave >> 1) * 32;   // wave row offset inside 64x64 tile
    const int wn   = (wave & 1)  * 32;   // wave col offset
    const int r    = lane & 15;
    const int hf   = lane >> 4;

    f32x8 acc[2][2];
#pragma unroll
    for (int i = 0; i < 2; i++)
#pragma unroll
        for (int j = 0; j < 2; j++)
            acc[i][j] = (f32x8){0.f,0.f,0.f,0.f,0.f,0.f,0.f,0.f};

    for (int k0 = 0; k0 < HSZ; k0 += 32) {
#pragma unroll
        for (int c = 0; c < 2; c++) {                 // 256 16B chunks / tile
            int idx = tid + c * 128;
            int row = idx >> 2;
            int ck  = (idx & 3) * 8;
            *(u32x4*)&As[row * 32 + ck] =
                *(const u32x4*)&A [(m0 + row) * HSZ + k0 + ck];
            *(u32x4*)&Bs[row * 32 + ck] =
                *(const u32x4*)&Bm[(n0 + row) * HSZ + k0 + ck];
        }
        __syncthreads();

        FragU a[2], b[2];
#pragma unroll
        for (int i = 0; i < 2; i++) {
            const unsigned short* pa = &As[(wm + i * 16 + r) * 32 + hf * 8];
            a[i].q[0] = *(const u32x4*)pa;
            a[i].q[1] = *(const u32x4*)(pa + 16);
            const unsigned short* pb = &Bs[(wn + i * 16 + r) * 32 + hf * 8];
            b[i].q[0] = *(const u32x4*)pb;
            b[i].q[1] = *(const u32x4*)(pb + 16);
        }
#pragma unroll
        for (int i = 0; i < 2; i++)
#pragma unroll
            for (int j = 0; j < 2; j++)
                acc[i][j] = __builtin_amdgcn_wmma_f32_16x16x32_bf16(
                    false, a[i].v, false, b[j].v, (short)0, acc[i][j],
                    false, false);
        __syncthreads();
    }

    // epilogue: C layout — lane gives n = lane&15, half = lane>>4; vgpr v -> m = v + 8*half
#pragma unroll
    for (int i = 0; i < 2; i++)
#pragma unroll
        for (int j = 0; j < 2; j++) {
            int ncol = n0 + wn + j * 16 + r;
            float bsv = bias[ncol];
#pragma unroll
            for (int v = 0; v < 8; v++) {
                int   mrow = m0 + wm + i * 16 + v + 8 * hf;
                float val  = acc[i][j][v] + bsv;
                if (mode == 0) {
                    ((float*)out)[(size_t)mrow * HSZ + ncol] = val;
                } else {
                    int bb = mrow >> 11, s = mrow & 2047;
                    int h  = ncol >> 6,  d = ncol & 63;
                    unsigned short bv16 = f32_to_bf16(val);
                    if (mode == 1)
                        ((unsigned short*)out)[((size_t)(bb * NH + h) * SEQ + s) * DH + d] = bv16;
                    else
                        ((unsigned short*)out)[((size_t)(bb * NH + h) * DH + d) * SEQ + s] = bv16;
                }
            }
        }
}

// -------------------------------------------------------- sigmoid attention ---
// Qh,Kh: [B*NH][SEQ][DH] bf16 ; Vt: [B*NH][DH][SEQ] bf16
// att:   [B*NH][SEQ][SEQ] f32 (part of d_out) ; attedb: [B][SEQ][HSZ] bf16 (ws)
__global__ __launch_bounds__(128)
void sig_attn_kernel(const unsigned short* __restrict__ Qh,
                     const unsigned short* __restrict__ Kh,
                     const unsigned short* __restrict__ Vt,
                     float* __restrict__ att,
                     unsigned short* __restrict__ attedb) {
    __shared__ unsigned short Qs[64 * 64];
    __shared__ unsigned short Ks[64 * 64];
    __shared__ unsigned short Vs[64 * 64];
    __shared__ unsigned short Ps[64 * 64];

    const int bh   = blockIdx.y;            // 0..31
    const int sq0  = blockIdx.x * 64;
    const int tid  = threadIdx.x;
    const int lane = tid & 31;
    const int wave = tid >> 5;
    const int wm   = (wave >> 1) * 32;
    const int wn   = (wave & 1)  * 32;
    const int r    = lane & 15;
    const int hf   = lane >> 4;

    const unsigned short* Qp = Qh + (size_t)bh * SEQ * DH;
    const unsigned short* Kp = Kh + (size_t)bh * SEQ * DH;
    const unsigned short* Vp = Vt + (size_t)bh * DH * SEQ;

#pragma unroll
    for (int c = 0; c < 4; c++) {           // Q tile: 512 16B chunks (sync path)
        int idx = tid + c * 128;
        int row = idx >> 3;
        int ck  = (idx & 7) * 8;
        *(u32x4*)&Qs[row * 64 + ck] = *(const u32x4*)&Qp[(sq0 + row) * DH + ck];
    }

    f32x8 oacc[2][2];
#pragma unroll
    for (int i = 0; i < 2; i++)
#pragma unroll
        for (int j = 0; j < 2; j++)
            oacc[i][j] = (f32x8){0.f,0.f,0.f,0.f,0.f,0.f,0.f,0.f};

    for (int t0 = 0; t0 < SEQ; t0 += 64) {
        // K/V tiles via CDNA5 async global->LDS DMA (no VGPR round-trip)
#pragma unroll
        for (int c = 0; c < 4; c++) {
            int idx = tid + c * 128;
            int row = idx >> 3;
            int ck  = (idx & 7) * 8;
            async_copy_b128(&Kp[(t0 + row) * DH + ck],  &Ks[row * 64 + ck]);
            async_copy_b128(&Vp[row * SEQ + t0 + ck],   &Vs[row * 64 + ck]);
            if (t0 + 64 < SEQ) {            // lowers to global_prefetch_b8
                __builtin_prefetch(&Kp[(t0 + 64 + row) * DH + ck], 0, 1);
                __builtin_prefetch(&Vp[row * SEQ + t0 + 64 + ck], 0, 1);
            }
        }
        wait_async_zero();                  // ASYNCcnt == 0: LDS writes landed
        __syncthreads();

        // scores = Q · K^T  (K dim = DH = 64 -> 2 WMMA k-steps)
#pragma unroll
        for (int i = 0; i < 2; i++)
#pragma unroll
            for (int j = 0; j < 2; j++) {
                f32x8 s = (f32x8){0.f,0.f,0.f,0.f,0.f,0.f,0.f,0.f};
#pragma unroll
                for (int kk = 0; kk < 2; kk++) {
                    FragU a, b;
                    const unsigned short* pa = &Qs[(wm + i * 16 + r) * 64 + kk * 32 + hf * 8];
                    a.q[0] = *(const u32x4*)pa;
                    a.q[1] = *(const u32x4*)(pa + 16);
                    const unsigned short* pb = &Ks[(wn + j * 16 + r) * 64 + kk * 32 + hf * 8];
                    b.q[0] = *(const u32x4*)pb;
                    b.q[1] = *(const u32x4*)(pb + 16);
                    s = __builtin_amdgcn_wmma_f32_16x16x32_bf16(
                        false, a.v, false, b.v, (short)0, s, false, false);
                }
                // scale by 1/sqrt(64), sigmoid, emit att_map + bf16 P tile
#pragma unroll
                for (int v = 0; v < 8; v++) {
                    float x = s[v] * 0.125f;
                    float p = 1.0f / (1.0f + __expf(-x));
                    int mrow = wm + i * 16 + v + 8 * hf;   // in 64x64 tile
                    int ncol = wn + j * 16 + r;
                    att[((size_t)bh * SEQ + sq0 + mrow) * SEQ + t0 + ncol] = p;
                    Ps[mrow * 64 + ncol] = f32_to_bf16(p);
                }
            }
        __syncthreads();

        // atted += P · V   (A = P rows, B = Vt rows: d-major, t-contiguous)
#pragma unroll
        for (int i = 0; i < 2; i++)
#pragma unroll
            for (int j = 0; j < 2; j++) {
#pragma unroll
                for (int kk = 0; kk < 2; kk++) {
                    FragU a, b;
                    const unsigned short* pa = &Ps[(wm + i * 16 + r) * 64 + kk * 32 + hf * 8];
                    a.q[0] = *(const u32x4*)pa;
                    a.q[1] = *(const u32x4*)(pa + 16);
                    const unsigned short* pb = &Vs[(wn + j * 16 + r) * 64 + kk * 32 + hf * 8];
                    b.q[0] = *(const u32x4*)pb;
                    b.q[1] = *(const u32x4*)(pb + 16);
                    oacc[i][j] = __builtin_amdgcn_wmma_f32_16x16x32_bf16(
                        false, a.v, false, b.v, (short)0, oacc[i][j],
                        false, false);
                }
            }
        __syncthreads();   // protect Ks/Vs/Ps before next iteration's loads
    }

    // write bf16 atted [B,SEQ,HSZ]
    const int b = bh >> 4, h = bh & 15;
#pragma unroll
    for (int i = 0; i < 2; i++)
#pragma unroll
        for (int j = 0; j < 2; j++)
#pragma unroll
            for (int v = 0; v < 8; v++) {
                int s   = sq0 + wm + i * 16 + v + 8 * hf;
                int col = h * DH + wn + j * 16 + r;
                attedb[((size_t)b * SEQ + s) * HSZ + col] = f32_to_bf16(oacc[i][j][v]);
            }
}

// ----------------------------------------------------------------- launch ---
extern "C" void kernel_launch(void* const* d_in, const int* in_sizes, int n_in,
                              void* d_out, int out_size, void* d_ws, size_t ws_size,
                              hipStream_t stream) {
    const float* v  = (const float*)d_in[0];
    const float* k  = (const float*)d_in[1];
    const float* q  = (const float*)d_in[2];
    const float* Wv = (const float*)d_in[3];
    const float* bv = (const float*)d_in[4];
    const float* Wk = (const float*)d_in[5];
    const float* bk = (const float*)d_in[6];
    const float* Wq = (const float*)d_in[7];
    const float* bq = (const float*)d_in[8];
    const float* Wm = (const float*)d_in[9];
    const float* bm = (const float*)d_in[10];

    float* out = (float*)d_out;                       // atted: 4096*1024 f32
    float* att = out + (size_t)MROWS * HSZ;           // att_map: 32*2048*2048 f32

    const size_t XH = (size_t)MROWS * HSZ;            // 4,194,304 elems
    const size_t WW = (size_t)HSZ * HSZ;              // 1,048,576 elems

    char* ws = (char*)d_ws;
    unsigned short* qb  = (unsigned short*)ws;  ws += XH * 2;
    unsigned short* kb  = (unsigned short*)ws;  ws += XH * 2;
    unsigned short* vb  = (unsigned short*)ws;  ws += XH * 2;
    unsigned short* Wqb = (unsigned short*)ws;  ws += WW * 2;
    unsigned short* Wkb = (unsigned short*)ws;  ws += WW * 2;
    unsigned short* Wvb = (unsigned short*)ws;  ws += WW * 2;
    unsigned short* Wmb = (unsigned short*)ws;  ws += WW * 2;
    unsigned short* QhP = (unsigned short*)ws;  ws += XH * 2;
    unsigned short* KhP = (unsigned short*)ws;  ws += XH * 2;
    unsigned short* VtP = (unsigned short*)ws;  ws += XH * 2;
    unsigned short* atb = (unsigned short*)ws;  ws += XH * 2;   // 64 MB total

    // f32 -> bf16 conversions
    cvt_bf16_kernel<<<2048, 256, 0, stream>>>(q,  qb,  (int)XH);
    cvt_bf16_kernel<<<2048, 256, 0, stream>>>(k,  kb,  (int)XH);
    cvt_bf16_kernel<<<2048, 256, 0, stream>>>(v,  vb,  (int)XH);
    cvt_bf16_kernel<<<1024, 256, 0, stream>>>(Wq, Wqb, (int)WW);
    cvt_bf16_kernel<<<1024, 256, 0, stream>>>(Wk, Wkb, (int)WW);
    cvt_bf16_kernel<<<1024, 256, 0, stream>>>(Wv, Wvb, (int)WW);
    cvt_bf16_kernel<<<1024, 256, 0, stream>>>(Wm, Wmb, (int)WW);

    dim3 ggrid(HSZ / 64, MROWS / 64);                 // (16, 64)
    gemm_bf16_kernel<<<ggrid, 128, 0, stream>>>(qb, Wqb, bq, QhP, 1);
    gemm_bf16_kernel<<<ggrid, 128, 0, stream>>>(kb, Wkb, bk, KhP, 1);
    gemm_bf16_kernel<<<ggrid, 128, 0, stream>>>(vb, Wvb, bv, VtP, 2);

    dim3 agrid(SEQ / 64, BATCH * NH);                 // (32, 32)
    sig_attn_kernel<<<agrid, 128, 0, stream>>>(QhP, KhP, VtP, att, atb);

    gemm_bf16_kernel<<<ggrid, 128, 0, stream>>>(atb, Wmb, bm, out, 0);
}
